// SPENModel_67903432950214
// MI455X (gfx1250) — compile-verified, matrix operand order checked by reading, not run on hardware
//
#include <hip/hip_runtime.h>
#include <math.h>

// ---------------------------------------------------------------------------
// SPEN model on MI455X (gfx1250): bf16 WMMA GEMMs + LDS-resident persistent
// inference loop.
//   xs (16384,1836) -> h=relu(xs@W1^T+b1) (16384,150) -> pots1 (16384,159)
//   29 mirror-descent iterations on pred (16384,159,2), then loss scalar.
// ---------------------------------------------------------------------------

#define BATCH   16384
#define INPUTS  1836
#define LAB     159
#define HID     150
#define NP      160       // 150/159 padded to 160 (10 WMMA n-tiles)
#define NIT     30
#define FEPS    1e-6f
#define MB      32        // rows per block in the inference kernel
#define KFULL   ((INPUTS / 64) * 64)   // 1792: last full-chunk boundary

typedef __attribute__((ext_vector_type(16))) __bf16 v16bf;
typedef __attribute__((ext_vector_type(8)))  float  v8f;

__device__ __forceinline__ v8f v8f_zero() {
  v8f z = {0.f, 0.f, 0.f, 0.f, 0.f, 0.f, 0.f, 0.f};
  return z;
}

__device__ __forceinline__ float frcp(float x) {
  return __builtin_amdgcn_rcpf(x);   // v_rcp_f32 (fast)
}

__device__ __forceinline__ v8f wmma_bf16(v16bf a, v16bf b, v8f c) {
  // D = A(16x32 bf16) x B(32x16 bf16) + C(16x16 f32)
  return __builtin_amdgcn_wmma_f32_16x16x32_bf16(false, a, false, b,
                                                 (short)0, c, false, false);
}

// 16 contiguous bf16 (32B, aligned) -> A/B fragment
__device__ __forceinline__ v16bf ld_frag_b(const __bf16* p) {
  return *(const v16bf*)p;
}

// 16 contiguous f32 from LDS, convert to bf16 fragment on the fly
__device__ __forceinline__ v16bf ld_frag_f32(const float* p) {
  v16bf a;
#pragma unroll
  for (int q = 0; q < 4; ++q) {
    float4 f = *(const float4*)(p + 4 * q);
    a[4*q+0] = (__bf16)f.x; a[4*q+1] = (__bf16)f.y;
    a[4*q+2] = (__bf16)f.z; a[4*q+3] = (__bf16)f.w;
  }
  return a;
}

// ---------------------------------------------------------------------------
// Kernel 1: h = relu(xs @ W1^T + b1), stored bf16, padded to 160 cols.
// Block = 64 rows x 160 cols, 8 waves; wave w -> mt = w>>1, nt = (w&1)+2*i.
// Main K-loop: 28 unguarded chunks of 64; one guarded remainder chunk (44).
// ---------------------------------------------------------------------------
struct __align__(32) Smem1 {
  __bf16 xsS[64 * 80];    // 64 rows x 64 K, stride 80 (row = 160B, 32B-mult)
  __bf16 w1S[160 * 80];   // 160 n x 64 K
};

// one K-chunk of WMMA work: grouped B loads -> back-to-back WMMA burst
__device__ __forceinline__ void feat1_mma(const Smem1& sm, int aBase, int ntb,
                                          int ln, int hk, v8f* acc) {
#pragma unroll
  for (int kk = 0; kk < 2; ++kk) {
    v16bf a = ld_frag_b(&sm.xsS[aBase + kk * 32]);
    v16bf bb[5];
#pragma unroll
    for (int i = 0; i < 5; ++i)
      bb[i] = ld_frag_b(
          &sm.w1S[((ntb + 2 * i) * 16 + ln) * 80 + hk * 16 + kk * 32]);
#pragma unroll
    for (int i = 0; i < 5; ++i) acc[i] = wmma_bf16(a, bb[i], acc[i]);
  }
}

__global__ __launch_bounds__(256) void k_feat1(const float* __restrict__ xs,
                                               const float* __restrict__ W1,
                                               const float* __restrict__ b1,
                                               __bf16* __restrict__ hb) {
  __shared__ Smem1 sm;
  const int tid = threadIdx.x;
  const int w = tid >> 5, lane = tid & 31, ln = lane & 15, hk = lane >> 4;
  const int row0 = blockIdx.x * 64;
  const int mt = w >> 1;          // 0..3
  const int ntb = w & 1;          // 0..1 ; nt = ntb + 2*i

  v8f acc[5];
#pragma unroll
  for (int i = 0; i < 5; ++i) acc[i] = v8f_zero();

  const int aBase = (mt * 16 + ln) * 80 + hk * 16;

  // staging coordinates (loop-invariant)
  const int xrow = tid >> 4, xc0 = (tid & 15) * 4;          // +r*16 rows
  const int wn = tid >> 4, wc0 = (tid & 15) * 4;            // +r*16 rows
  const float* xbase = xs + (size_t)(row0 + xrow) * INPUTS + xc0;

  // ---- main loop: full 64-wide chunks, no guards ----
  for (int kb = 0; kb < KFULL; kb += 64) {
#pragma unroll
    for (int r = 0; r < 4; ++r) {
      float4 f = *(const float4*)(xbase + (size_t)r * 16 * INPUTS + kb);
      __bf16* d = &sm.xsS[(xrow + r * 16) * 80 + xc0];
      d[0] = (__bf16)f.x; d[1] = (__bf16)f.y;
      d[2] = (__bf16)f.z; d[3] = (__bf16)f.w;
    }
#pragma unroll
    for (int r = 0; r < 10; ++r) {
      int n = wn + r * 16;
      int nc = (n < HID) ? n : (HID - 1);                    // clamped row
      float msk = (n < HID) ? 1.f : 0.f;                     // branchless mask
      float4 f = *(const float4*)(W1 + (size_t)nc * INPUTS + kb + wc0);
      __bf16* d = &sm.w1S[n * 80 + wc0];
      d[0] = (__bf16)(f.x * msk); d[1] = (__bf16)(f.y * msk);
      d[2] = (__bf16)(f.z * msk); d[3] = (__bf16)(f.w * msk);
    }
    // prefetch next xs chunk
    __builtin_prefetch(xs + (size_t)(row0 + (tid & 63)) * INPUTS + kb + 64 +
                           (tid >> 6) * 16, 0, 0);
    __syncthreads();
    feat1_mma(sm, aBase, ntb, ln, hk, acc);
    __syncthreads();
  }

  // ---- remainder chunk (cols 1792..1835), guarded, runs once ----
  {
    const int kb = KFULL;
#pragma unroll
    for (int r = 0; r < 4; ++r) {
      const float* gp = xbase + (size_t)r * 16 * INPUTS + kb;
      float4 f;
      f.x = (kb + xc0 + 0 < INPUTS) ? gp[0] : 0.f;
      f.y = (kb + xc0 + 1 < INPUTS) ? gp[1] : 0.f;
      f.z = (kb + xc0 + 2 < INPUTS) ? gp[2] : 0.f;
      f.w = (kb + xc0 + 3 < INPUTS) ? gp[3] : 0.f;
      __bf16* d = &sm.xsS[(xrow + r * 16) * 80 + xc0];
      d[0] = (__bf16)f.x; d[1] = (__bf16)f.y;
      d[2] = (__bf16)f.z; d[3] = (__bf16)f.w;
    }
#pragma unroll
    for (int r = 0; r < 10; ++r) {
      int n = wn + r * 16;
      int nc = (n < HID) ? n : (HID - 1);
      float msk = (n < HID) ? 1.f : 0.f;
      const float* gp = W1 + (size_t)nc * INPUTS + kb + wc0;
      float4 f;
      f.x = (kb + wc0 + 0 < INPUTS) ? gp[0] * msk : 0.f;
      f.y = (kb + wc0 + 1 < INPUTS) ? gp[1] * msk : 0.f;
      f.z = (kb + wc0 + 2 < INPUTS) ? gp[2] * msk : 0.f;
      f.w = (kb + wc0 + 3 < INPUTS) ? gp[3] * msk : 0.f;
      __bf16* d = &sm.w1S[n * 80 + wc0];
      d[0] = (__bf16)f.x; d[1] = (__bf16)f.y;
      d[2] = (__bf16)f.z; d[3] = (__bf16)f.w;
    }
    __syncthreads();
    feat1_mma(sm, aBase, ntb, ln, hk, acc);
  }

  // epilogue: bias + relu, store bf16 (padded cols -> 0)
#pragma unroll
  for (int i = 0; i < 5; ++i) {
    int nt = ntb + 2 * i;
    int col = nt * 16 + ln;
    float bias = (col < HID) ? b1[col] : 0.f;
#pragma unroll
    for (int v = 0; v < 8; ++v) {
      int row = row0 + mt * 16 + v + 8 * hk;
      float h = (col < HID) ? fmaxf(acc[i][v] + bias, 0.f) : 0.f;
      hb[(size_t)row * NP + col] = (__bf16)h;
    }
  }
}

// ---------------------------------------------------------------------------
// Kernel 2: pots1 = h @ W2^T + b2 (16384 x 159, padded to 160, f32)
// ---------------------------------------------------------------------------
struct __align__(32) Smem2 {
  __bf16 hbS[64 * NP];
  __bf16 w2S[NP * NP];   // [n][k]
};

__global__ __launch_bounds__(256) void k_feat2(const __bf16* __restrict__ hb,
                                               const float* __restrict__ W2,
                                               const float* __restrict__ b2,
                                               float* __restrict__ pots1) {
  __shared__ Smem2 sm;
  const int tid = threadIdx.x;
  const int w = tid >> 5, lane = tid & 31, ln = lane & 15, hk = lane >> 4;
  const int row0 = blockIdx.x * 64;
  const int mt = w >> 1;
  const int ntb = w & 1;

  {  // stage h tile (bf16 copy, 16B vectors)
    const uint4* src = (const uint4*)(hb + (size_t)row0 * NP);
    uint4* dst = (uint4*)sm.hbS;
    for (int i = tid; i < (64 * NP) / 8; i += 256) dst[i] = src[i];
  }
  // stage W2 -> w2S[n][k] bf16 (n>=159 or k>=150 -> 0)
  for (int idx = tid; idx < NP * NP; idx += 256) {
    int n = idx / NP, k = idx - n * NP;
    float v = (n < LAB && k < HID) ? W2[n * HID + k] : 0.f;
    sm.w2S[idx] = (__bf16)v;
  }
  __syncthreads();

  const int aBase = (mt * 16 + ln) * NP + hk * 16;
  v8f acc[5];
#pragma unroll
  for (int i = 0; i < 5; ++i) acc[i] = v8f_zero();

#pragma unroll
  for (int kk = 0; kk < 5; ++kk) {
    v16bf a = ld_frag_b(&sm.hbS[aBase + kk * 32]);
    v16bf bb[5];
#pragma unroll
    for (int i = 0; i < 5; ++i)
      bb[i] = ld_frag_b(
          &sm.w2S[((ntb + 2 * i) * 16 + ln) * NP + hk * 16 + kk * 32]);
#pragma unroll
    for (int i = 0; i < 5; ++i) acc[i] = wmma_bf16(a, bb[i], acc[i]);
  }

#pragma unroll
  for (int i = 0; i < 5; ++i) {
    int nt = ntb + 2 * i;
    int col = nt * 16 + ln;
    float bias = (col < LAB) ? b2[col] : 0.f;
#pragma unroll
    for (int v = 0; v < 8; ++v) {
      int row = row0 + mt * 16 + v + 8 * hk;
      float o = (col < LAB) ? acc[i][v] + bias : 0.f;
      pots1[(size_t)row * NP + col] = o;
    }
  }
}

// ---------------------------------------------------------------------------
// Kernel 3: persistent LDS-resident inference loop + fused loss partials.
// Per block: 32 rows, 256 threads (8 waves), ~152KB LDS.
// Wave w -> mt = w>>2, nt = (w&3)+4*i (i<3, wave-uniform guard nt<10).
// ---------------------------------------------------------------------------
struct __align__(32) Smem3 {
  __bf16 wg1a[NP * NP];   // [t][l]  (B for Z = y1 @ Wg1^T)
  __bf16 wg1b[NP * NP];   // [l][t]  (B for g = s @ Wg1)
  __bf16 sB[MB * NP];     // [b][t]  sigmoid(Z)*Wg2, bf16
  float  p0[MB * NP];
  float  p1[MB * NP];
  float  wg2[NP];
  float  part[MB * 10];
  float  rowA[MB];
};

// Z = p1 @ Wg1^T for this wave's tiles; A fragment hoisted across N-tiles.
__device__ __forceinline__ void z_mma(const float* p1v, const __bf16* wg1a,
                                      int aBase, int ntb, int ln, int hk,
                                      v8f* acc3) {
#pragma unroll
  for (int i = 0; i < 3; ++i) acc3[i] = v8f_zero();
#pragma unroll
  for (int kk = 0; kk < 5; ++kk) {
    v16bf a = ld_frag_f32(&p1v[aBase + kk * 32]);
    v16bf bb[3];
#pragma unroll
    for (int i = 0; i < 3; ++i) {
      int nt = ntb + 4 * i;
      if (nt < 10)
        bb[i] = ld_frag_b(&wg1a[(nt * 16 + ln) * NP + hk * 16 + kk * 32]);
    }
#pragma unroll
    for (int i = 0; i < 3; ++i) {
      int nt = ntb + 4 * i;
      if (nt < 10) acc3[i] = wmma_bf16(a, bb[i], acc3[i]);
    }
  }
}

// label_b = sum_t wg2[t]*softplus(Z[b,t]); deterministic shuffle reduce.
__device__ __forceinline__ void z_label_partials(const float* p1v,
                                                 const __bf16* wg1a,
                                                 const float* wg2, float* part,
                                                 int mt, int ntb, int ln,
                                                 int hk) {
  const int aBase = (mt * 16 + ln) * NP + hk * 16;
  v8f acc3[3];
  z_mma(p1v, wg1a, aBase, ntb, ln, hk, acc3);
#pragma unroll
  for (int i = 0; i < 3; ++i) {
    int nt = ntb + 4 * i;
    if (nt < 10) {
      int t = nt * 16 + ln;
      float wt = (t < HID) ? wg2[t] : 0.f;
#pragma unroll
      for (int v = 0; v < 8; ++v) {
        int bq = mt * 16 + v + 8 * hk;
        float z = acc3[i][v];
        float spv = (z > 0.f) ? (z + log1pf(__expf(-z))) : log1pf(__expf(z));
        float sp = wt * spv;
        sp += __shfl_xor(sp, 1);
        sp += __shfl_xor(sp, 2);
        sp += __shfl_xor(sp, 4);
        sp += __shfl_xor(sp, 8);
        if (ln == 0) part[bq * 10 + nt] = sp;
      }
    }
  }
}

__global__ __launch_bounds__(256) void k_infer(const float* __restrict__ pots1,
                                               const float* __restrict__ pred0,
                                               const float* __restrict__ Wg1,
                                               const float* __restrict__ Wg2,
                                               const float* __restrict__ ys,
                                               float* __restrict__ partials) {
  __shared__ Smem3 sm;
  const int tid = threadIdx.x;
  const int w = tid >> 5, lane = tid & 31, ln = lane & 15, hk = lane >> 4;
  const int row0 = blockIdx.x * MB;
  const int mt = w >> 2;          // 0..1
  const int ntb = w & 3;          // 0..3 ; nt = ntb + 4*i

  // ---- init LDS: Wg1 (both layouts, zero-padded), Wg2, pred state ----
  for (int idx = tid; idx < NP * NP; idx += 256) {
    int t = idx / NP, l = idx - t * NP;
    float v = (t < HID && l < LAB) ? Wg1[t * LAB + l] : 0.f;
    sm.wg1a[t * NP + l] = (__bf16)v;
    sm.wg1b[l * NP + t] = (__bf16)v;
  }
  for (int t = tid; t < NP; t += 256) sm.wg2[t] = (t < HID) ? Wg2[t] : 0.f;
  for (int idx = tid; idx < MB * NP; idx += 256) {
    int l = idx % NP;
    float q0 = 0.f, q1 = 0.f;
    if (l < LAB) { q0 = pred0[2 * l]; q1 = pred0[2 * l + 1]; }
    sm.p0[idx] = q0;
    sm.p1[idx] = q1;
  }
  __syncthreads();

  const int aBase = (mt * 16 + ln) * NP + hk * 16;

  // ---- 29 mirror-descent iterations, all state LDS-resident ----
  for (int it = 1; it < NIT; ++it) {
    float lr = 0.05f / sqrtf((float)it);

    // Z = p1 @ Wg1^T ; sB = sigmoid(Z) * Wg2  (bf16)
    {
      v8f acc3[3];
      z_mma(sm.p1, sm.wg1a, aBase, ntb, ln, hk, acc3);
#pragma unroll
      for (int i = 0; i < 3; ++i) {
        int nt = ntb + 4 * i;
        if (nt < 10) {
          int t = nt * 16 + ln;
          float wt = (t < HID) ? sm.wg2[t] : 0.f;
#pragma unroll
          for (int v = 0; v < 8; ++v) {
            int bq = mt * 16 + v + 8 * hk;
            float s = wt * frcp(1.f + __expf(-acc3[i][v]));
            sm.sB[bq * NP + t] = (__bf16)s;
          }
        }
      }
    }
    __syncthreads();

    // g1 = sB @ Wg1 ; multiplicative update, done in D-fragment layout
    {
      v8f acc3[3];
#pragma unroll
      for (int i = 0; i < 3; ++i) acc3[i] = v8f_zero();
#pragma unroll
      for (int kk = 0; kk < 5; ++kk) {
        v16bf a = ld_frag_b(&sm.sB[aBase + kk * 32]);
        v16bf bb[3];
#pragma unroll
        for (int i = 0; i < 3; ++i) {
          int nt = ntb + 4 * i;
          if (nt < 10)
            bb[i] =
                ld_frag_b(&sm.wg1b[(nt * 16 + ln) * NP + hk * 16 + kk * 32]);
        }
#pragma unroll
        for (int i = 0; i < 3; ++i) {
          int nt = ntb + 4 * i;
          if (nt < 10) acc3[i] = wmma_bf16(a, bb[i], acc3[i]);
        }
      }
#pragma unroll
      for (int i = 0; i < 3; ++i) {
        int nt = ntb + 4 * i;
        if (nt < 10) {
          int l = nt * 16 + ln;
          if (l < LAB) {
            const float* potcol =
                pots1 + (size_t)(row0 + mt * 16 + 8 * hk) * NP + l;
#pragma unroll
            for (int v = 0; v < 8; ++v) {
              int bq = mt * 16 + v + 8 * hk;
              int o = bq * NP + l;
              float P0 = sm.p0[o], P1 = sm.p1[o];
              float pt = potcol[(size_t)v * NP];  // L2-resident
              float g0 = -(__logf(P0 + FEPS) + P0 * frcp(P0 + FEPS));
              float g1 =
                  pt + acc3[i][v] - (__logf(P1 + FEPS) + P1 * frcp(P1 + FEPS));
              float r0 = lr * g0, r1 = lr * g1;
              float mx = fmaxf(r0, r1);
              float e0 = P0 * __expf(r0 - mx);
              float e1 = P1 * __expf(r1 - mx);
              float inv = frcp(e0 + e1 + FEPS);
              sm.p0[o] = e0 * inv;
              sm.p1[o] = e1 * inv;
            }
          }
        }
      }
    }
    __syncthreads();
  }

  // ---- loss: (local_p + label_p) - (local_t + label_t) - entropy(pred) ----
  z_label_partials(sm.p1, sm.wg1a, sm.wg2, sm.part, mt, ntb, ln, hk);
  __syncthreads();
  if (tid < MB) {
    float labelP = 0.f;
#pragma unroll
    for (int nt = 0; nt < 10; ++nt) labelP += sm.part[tid * 10 + nt];
    float localP = 0.f, entP = 0.f;
    for (int l = 0; l < LAB; ++l) {
      float P0 = sm.p0[tid * NP + l], P1 = sm.p1[tid * NP + l];
      localP += P1 * pots1[(size_t)(row0 + tid) * NP + l];
      entP += P0 * __logf(P0 + FEPS) + P1 * __logf(P1 + FEPS);
    }
    sm.rowA[tid] = localP + labelP - entP;
  }
  __syncthreads();

  // overwrite p1 with ys component-1 for the true-energy pass
  for (int idx = tid; idx < MB * NP; idx += 256) {
    int b = idx / NP, l = idx - b * NP;
    sm.p1[idx] = (l < LAB) ? ys[(size_t)(row0 + b) * (2 * LAB) + 2 * l + 1]
                           : 0.f;
  }
  __syncthreads();
  z_label_partials(sm.p1, sm.wg1a, sm.wg2, sm.part, mt, ntb, ln, hk);
  __syncthreads();
  if (tid < MB) {
    float labelT = 0.f;
#pragma unroll
    for (int nt = 0; nt < 10; ++nt) labelT += sm.part[tid * 10 + nt];
    float localT = 0.f;
    for (int l = 0; l < LAB; ++l)
      localT += sm.p1[tid * NP + l] * pots1[(size_t)(row0 + tid) * NP + l];
    sm.rowA[tid] -= (localT + labelT);
  }
  __syncthreads();
  if (tid == 0) {
    float s = 0.f;
    for (int r = 0; r < MB; ++r) s += sm.rowA[r];  // fixed order: deterministic
    partials[blockIdx.x] = s;
  }
}

// ---------------------------------------------------------------------------
// Kernel 4: deterministic fixed-order sum of block partials -> mean loss.
// ---------------------------------------------------------------------------
__global__ void k_final(const float* __restrict__ partials,
                        float* __restrict__ out, int n) {
  if (threadIdx.x == 0 && blockIdx.x == 0) {
    double s = 0.0;
    for (int i = 0; i < n; ++i) s += (double)partials[i];
    out[0] = (float)(s / (double)BATCH);
  }
}

// ---------------------------------------------------------------------------
extern "C" void kernel_launch(void* const* d_in, const int* in_sizes, int n_in,
                              void* d_out, int out_size, void* d_ws,
                              size_t ws_size, hipStream_t stream) {
  (void)in_sizes; (void)n_in; (void)out_size; (void)ws_size;
  const float* xs    = (const float*)d_in[0];
  const float* ys    = (const float*)d_in[1];
  const float* pred0 = (const float*)d_in[2];
  const float* W1    = (const float*)d_in[3];
  const float* b1    = (const float*)d_in[4];
  const float* W2    = (const float*)d_in[5];
  const float* b2    = (const float*)d_in[6];
  const float* Wg1   = (const float*)d_in[7];
  const float* Wg2   = (const float*)d_in[8];

  char* ws = (char*)d_ws;
  float*  pots1 = (float*)ws;                                   // BATCH*NP f32
  __bf16* hb    = (__bf16*)(ws + (size_t)BATCH * NP * 4);       // BATCH*NP bf16
  float*  parts = (float*)(ws + (size_t)BATCH * NP * 4 +
                                (size_t)BATCH * NP * 2);        // BATCH/MB f32

  k_feat1<<<BATCH / 64, 256, 0, stream>>>(xs, W1, b1, hb);
  k_feat2<<<BATCH / 64, 256, 0, stream>>>(hb, W2, b2, pots1);
  k_infer<<<BATCH / MB, 256, 0, stream>>>(pots1, pred0, Wg1, Wg2, ys, parts);
  k_final<<<1, 32, 0, stream>>>(parts, (float*)d_out, BATCH / MB);
}
